// HiDeAdapterModel_78460462563329
// MI455X (gfx1250) — matrix operand order repaired and checked by reading, not compiled, not run on hardware
//
#include <hip/hip_runtime.h>

// ---------------------------------------------------------------------------
// HiDeAdapter ViT forward for MI455X (gfx1250, wave32, WMMA).
// GEMM-dominated (~1.07 TFLOP): v_wmma_f32_16x16x32_bf16 for all big matmuls.
// Weights are converted fp32->bf16 AND transposed once ([K][N] -> [N][K]) so
// both GEMM operands stage into LDS with contiguous 16B copies and fragments
// are pure ds_load_b128 pairs. 128x64 block tile, 32x32 per wave (4 WMMA /
// K-step), ping-pong LDS buffers, one barrier per K-step.
// ---------------------------------------------------------------------------

typedef __attribute__((ext_vector_type(16))) __bf16 v16bf;
typedef __attribute__((ext_vector_type(8)))  __bf16 v8bf;
typedef __attribute__((ext_vector_type(8)))  float  v8f;

#define Bc  32
#define Nn  197
#define Dd  768
#define Hh  12
#define DH  64
#define BNr (Bc * Nn)      // 6304 token rows
#define GG  196            // 14*14 patches
#define Cc  100

static __device__ __forceinline__ __bf16 f2bf(float f) {
  union { float f; unsigned u; } c; c.f = f;
  unsigned u = c.u;
  u += 0x7FFFu + ((u >> 16) & 1u);           // round-to-nearest-even
  union { unsigned short s; __bf16 b; } o; o.s = (unsigned short)(u >> 16);
  return o.b;
}

// ---------------------------------------------------------------------------
// bf16 WMMA GEMM: out[M,N] = A[M,K] * BT[N,K]^T + bias (+ res)
//   A  : bf16 row-major [M][K]   (activations)
//   BT : bf16 row-major [N][K]   (pre-transposed weights)
// Block 256 threads = 8 waves; tile 128(M) x 64(N); K-step 32.
// Wave w: rows ((w>>1)*32), cols ((w&1)*32): 2 A-frags x 2 B-frags, 4 WMMAs.
// A fragment (ISA 16-bit A 16x32): lane(half,lr) reads row wm+lr, two 16B
//   chunks at K = half*8 and 16+half*8.
// B fragment (ISA 16-bit B 32x16): lane(half,lr) reads BT row wn+lr, two
//   16B chunks at K = half*8 and 16+half*8.
// C/D: VGPR r -> M = r + half*8, N = lr.
// All call sites have N % 64 == 0 and K % 32 == 0; only M needs guarding.
// ---------------------------------------------------------------------------
#define TM 128
#define TN 64
#define STR 40   // padded LDS row stride in bf16 elems (80B: keeps 16B align)

template <bool GUARD>
__global__ __launch_bounds__(256)
void k_gemm_bf16(const __bf16* __restrict__ A, const __bf16* __restrict__ BT,
                 const float* __restrict__ bias, const float* __restrict__ res,
                 float* __restrict__ out, int M, int N, int K, int m0_base)
{
  __shared__ __bf16 As[2][TM * STR];
  __shared__ __bf16 Bs[2][TN * STR];

  const int tid  = threadIdx.x;
  const int wid  = tid >> 5;
  const int lane = tid & 31;
  const int m0 = m0_base + blockIdx.y * TM;
  const int n0 = blockIdx.x * TN;
  const int wm = (wid >> 1) * 32;
  const int wn = (wid & 1) * 32;
  const int half = lane >> 4;
  const int lr   = lane & 15;

  // staging coordinates
  const int arow = tid >> 1;            // 0..127
  const int aseg = (tid & 1) * 16;      // 0 or 16 elems
  const int brow = tid >> 2;            // 0..63
  const int bseg = (tid & 3) * 8;       // 0,8,16,24 elems

  auto stage = [&](int kk, int bsel) {
    // A: 128 x 32, contiguous 32B per thread
    {
      __bf16* dst = &As[bsel][arow * STR + aseg];
      bool ok = true;
      if (GUARD) ok = (m0 + arow) < M;
      if (ok) {
        const v8bf* p = (const v8bf*)(A + (size_t)(m0 + arow) * K + kk + aseg);
        v8bf u0 = p[0], u1 = p[1];
        *(v8bf*)dst = u0; *(v8bf*)(dst + 8) = u1;
      } else {
        v8bf z = {};
        *(v8bf*)dst = z; *(v8bf*)(dst + 8) = z;
      }
    }
    // B: 64 x 32, contiguous 16B per thread
    {
      const v8bf* p = (const v8bf*)(BT + (size_t)(n0 + brow) * K + kk + bseg);
      *(v8bf*)&Bs[bsel][brow * STR + bseg] = p[0];
    }
  };

  v8f acc00 = {}, acc01 = {}, acc10 = {}, acc11 = {};

  stage(0, 0);
  __syncthreads();

  int buf = 0;
  for (int k0 = 0; k0 < K; k0 += 32) {
    if (k0 + 32 < K) stage(k0 + 32, buf ^ 1);

    const __bf16* Ab = As[buf];
    const __bf16* Bb = Bs[buf];

    const __bf16* ap0 = &Ab[(wm + lr) * STR];
    const __bf16* ap1 = &Ab[(wm + 16 + lr) * STR];
    v8bf a00 = *(const v8bf*)(ap0 + half * 8);
    v8bf a01 = *(const v8bf*)(ap0 + 16 + half * 8);
    v8bf a10 = *(const v8bf*)(ap1 + half * 8);
    v8bf a11 = *(const v8bf*)(ap1 + 16 + half * 8);

    const __bf16* bp0 = &Bb[(wn + lr) * STR];
    const __bf16* bp1 = &Bb[(wn + 16 + lr) * STR];
    v8bf b00 = *(const v8bf*)(bp0 + half * 8), b01 = *(const v8bf*)(bp0 + 16 + half * 8);
    v8bf b10 = *(const v8bf*)(bp1 + half * 8), b11 = *(const v8bf*)(bp1 + 16 + half * 8);

    v16bf a0, a1, b0, b1;
    #pragma unroll
    for (int j = 0; j < 8; ++j) {
      a0[j] = a00[j]; a0[j + 8] = a01[j];
      a1[j] = a10[j]; a1[j + 8] = a11[j];
      b0[j] = b00[j]; b0[j + 8] = b01[j];
      b1[j] = b10[j]; b1[j + 8] = b11[j];
    }

    acc00 = __builtin_amdgcn_wmma_f32_16x16x32_bf16(false, a0, false, b0, (short)0, acc00, false, false);
    acc01 = __builtin_amdgcn_wmma_f32_16x16x32_bf16(false, a0, false, b1, (short)0, acc01, false, false);
    acc10 = __builtin_amdgcn_wmma_f32_16x16x32_bf16(false, a1, false, b0, (short)0, acc10, false, false);
    acc11 = __builtin_amdgcn_wmma_f32_16x16x32_bf16(false, a1, false, b1, (short)0, acc11, false, false);

    __syncthreads();
    buf ^= 1;
  }

  // ---- epilogue ----
  const int c0 = n0 + wn + lr;
  const int c1 = c0 + 16;
  const float bia0 = bias ? bias[c0] : 0.f;
  const float bia1 = bias ? bias[c1] : 0.f;
  #pragma unroll
  for (int r = 0; r < 8; ++r) {
    const int mr0 = m0 + wm + r + half * 8;
    const int mr1 = mr0 + 16;
    if (!GUARD || mr0 < M) {
      const size_t ro = (size_t)mr0 * N;
      out[ro + c0] = acc00[r] + bia0 + (res ? res[ro + c0] : 0.f);
      out[ro + c1] = acc01[r] + bia1 + (res ? res[ro + c1] : 0.f);
    }
    if (!GUARD || mr1 < M) {
      const size_t ro = (size_t)mr1 * N;
      out[ro + c0] = acc10[r] + bia0 + (res ? res[ro + c0] : 0.f);
      out[ro + c1] = acc11[r] + bia1 + (res ? res[ro + c1] : 0.f);
    }
  }
}

// ---------------------------------------------------------------------------
// fp32 [K][N] -> bf16 transposed [N][K], tiled 32x32 through LDS.
// ---------------------------------------------------------------------------
__global__ __launch_bounds__(256)
void k_f2bfT(const float* __restrict__ in, __bf16* __restrict__ o, int K, int N)
{
  __shared__ __bf16 tile[32][33];
  const int k0 = blockIdx.y * 32;
  const int n0 = blockIdx.x * 32;
  const int tx = threadIdx.x & 31;
  const int ty = threadIdx.x >> 5;   // 0..7
  #pragma unroll
  for (int j = 0; j < 32; j += 8)
    tile[ty + j][tx] = f2bf(in[(size_t)(k0 + ty + j) * N + n0 + tx]);
  __syncthreads();
  #pragma unroll
  for (int j = 0; j < 32; j += 8)
    o[(size_t)(n0 + ty + j) * K + k0 + tx] = tile[tx][ty + j];
}

// ---------------------------------------------------------------------------
// im2col patch extraction:  x(B,3,224,224) -> bf16 [B*196, 768]
// ---------------------------------------------------------------------------
__global__ void k_im2col(const float* __restrict__ x, __bf16* __restrict__ o)
{
  const size_t i = (size_t)blockIdx.x * 256 + threadIdx.x;
  if (i >= (size_t)Bc * GG * Dd) return;
  const int col = (int)(i % Dd);
  const size_t r = i / Dd;
  const int gg = (int)(r % GG);
  const int b  = (int)(r / GG);
  const int gy = gg / 14, gx = gg % 14;
  const int c  = col >> 8;
  const int rem = col & 255;
  const int py = rem >> 4, px = rem & 15;
  const float v = x[(((size_t)b * 3 + c) * 224 + gy * 16 + py) * 224 + gx * 16 + px];
  o[i] = f2bf(v);
}

// h[b,0,:] = cls + pos[0];  h[b,1+n,:] = pe[b,n,:] + pos[1+n]
__global__ void k_assemble(const float* __restrict__ pe, const float* __restrict__ cls,
                           const float* __restrict__ pos, float* __restrict__ h)
{
  const size_t i = (size_t)blockIdx.x * 256 + threadIdx.x;
  if (i >= (size_t)BNr * Dd) return;
  const int d = (int)(i % Dd);
  const size_t r = i / Dd;
  const int nn = (int)(r % Nn);
  const int b  = (int)(r / Nn);
  float v;
  if (nn == 0) v = cls[d] + pos[d];
  else v = pe[((size_t)b * GG + nn - 1) * Dd + d] + pos[(size_t)nn * Dd + d];
  h[i] = v;
}

__global__ void k_f2bf(const float* __restrict__ in, __bf16* __restrict__ o, int n)
{
  const int i = blockIdx.x * 256 + threadIdx.x;
  if (i < n) o[i] = f2bf(in[i]);
}

__global__ void k_gelu_bf16(const float* __restrict__ in, __bf16* __restrict__ o, int n)
{
  const int i = blockIdx.x * 256 + threadIdx.x;
  if (i < n) {
    const float x = in[i];
    o[i] = f2bf(0.5f * x * (1.f + erff(x * 0.70710678118654752f)));
  }
}

__global__ void k_resid(float* __restrict__ h, const float* __restrict__ m, int n)
{
  const int i = blockIdx.x * 256 + threadIdx.x;
  if (i < n) h[i] += m[i];
}

// LayerNorm over D=768, output bf16 (GEMM A input). One block per token row.
__global__ __launch_bounds__(256)
void k_ln_bf16(const float* __restrict__ h, const float* __restrict__ s,
               const float* __restrict__ bb, __bf16* __restrict__ ob)
{
  const int bn = blockIdx.x;
  const int t  = threadIdx.x;
  __shared__ float red[256];
  const float* r = h + (size_t)bn * Dd;

  float sum = 0.f;
  for (int d = t; d < Dd; d += 256) sum += r[d];
  red[t] = sum; __syncthreads();
  for (int o = 128; o > 0; o >>= 1) { if (t < o) red[t] += red[t + o]; __syncthreads(); }
  const float mean = red[0] * (1.f / Dd); __syncthreads();

  float var = 0.f;
  for (int d = t; d < Dd; d += 256) { const float x = r[d] - mean; var += x * x; }
  red[t] = var; __syncthreads();
  for (int o = 128; o > 0; o >>= 1) { if (t < o) red[t] += red[t + o]; __syncthreads(); }
  const float rstd = rsqrtf(red[0] * (1.f / Dd) + 1e-6f);

  for (int d = t; d < Dd; d += 256)
    ob[(size_t)bn * Dd + d] = f2bf((r[d] - mean) * rstd * s[d] + bb[d]);
}

// Attention: one block per (query n, batch b); loops all 12 heads.
// qkv row layout: [3][H][64]  (q@0, k@768, v@1536)
__global__ __launch_bounds__(128)
void k_attn(const float* __restrict__ qkv, float* __restrict__ o)
{
  const int n = blockIdx.x;
  const int b = blockIdx.y;
  const int t = threadIdx.x;
  __shared__ float q[DH];
  __shared__ float sc[208];
  __shared__ float red[128];
  const size_t rowQ = ((size_t)b * Nn + n) * 2304;

  for (int hh = 0; hh < Hh; ++hh) {
    if (t < DH) q[t] = qkv[rowQ + hh * DH + t];
    __syncthreads();

    float lmax = -1e30f;
    for (int j = t; j < Nn; j += 128) {
      const float* kr = qkv + ((size_t)b * Nn + j) * 2304 + 768 + hh * DH;
      float s = 0.f;
      for (int d = 0; d < DH; ++d) s += q[d] * kr[d];
      s *= 0.125f;                      // 64^-0.5
      sc[j] = s;
      lmax = fmaxf(lmax, s);
    }
    red[t] = lmax; __syncthreads();
    for (int off = 64; off > 0; off >>= 1) { if (t < off) red[t] = fmaxf(red[t], red[t + off]); __syncthreads(); }
    const float m = red[0]; __syncthreads();

    float lsum = 0.f;
    for (int j = t; j < Nn; j += 128) { const float e = expf(sc[j] - m); sc[j] = e; lsum += e; }
    red[t] = lsum; __syncthreads();
    for (int off = 64; off > 0; off >>= 1) { if (t < off) red[t] += red[t + off]; __syncthreads(); }
    const float inv = 1.f / red[0]; __syncthreads();

    if (t < DH) {
      float acc = 0.f;
      for (int j = 0; j < Nn; ++j)
        acc += sc[j] * qkv[((size_t)b * Nn + j) * 2304 + 1536 + hh * DH + t];
      o[((size_t)b * Nn + n) * Dd + hh * DH + t] = acc * inv;
    }
    __syncthreads();
  }
}

// Adapter + residual:  h = h + relu(h@down + db)@up + ub + mlp
__global__ __launch_bounds__(256)
void k_adapter(float* __restrict__ h, const float* __restrict__ mlp,
               const int* __restrict__ tids,
               const float* __restrict__ dw, const float* __restrict__ db,
               const float* __restrict__ uw, const float* __restrict__ ub)
{
  const int bn = blockIdx.x;
  const int b  = bn / Nn;
  const int t  = threadIdx.x;
  const int e  = tids[b];
  __shared__ float red[256];
  __shared__ float a8[8];
  const float* hr = h + (size_t)bn * Dd;
  const float* dwe = dw + (size_t)e * Dd * 8;

  float acc[8] = {0.f, 0.f, 0.f, 0.f, 0.f, 0.f, 0.f, 0.f};
  for (int d = t; d < Dd; d += 256) {
    const float hv = hr[d];
    const float* p = dwe + d * 8;
    #pragma unroll
    for (int k = 0; k < 8; ++k) acc[k] += hv * p[k];
  }
  for (int k = 0; k < 8; ++k) {
    red[t] = acc[k]; __syncthreads();
    for (int off = 128; off > 0; off >>= 1) { if (t < off) red[t] += red[t + off]; __syncthreads(); }
    if (t == 0) a8[k] = fmaxf(red[0] + db[e * 8 + k], 0.f);
    __syncthreads();
  }
  const float* uwe = uw + (size_t)e * 8 * Dd;
  const float* ube = ub + (size_t)e * Dd;
  for (int d = t; d < Dd; d += 256) {
    float ad = ube[d];
    #pragma unroll
    for (int k = 0; k < 8; ++k) ad += a8[k] * uwe[k * Dd + d];
    h[(size_t)bn * Dd + d] = hr[d] + mlp[(size_t)bn * Dd + d] + ad;
  }
}

// Final norm (row 0 per batch) -> feat, then logits = feat @ head_w + head_b.
// d_out: [B*C logits][B*D feat]
__global__ __launch_bounds__(256)
void k_head(const float* __restrict__ h, const float* __restrict__ ns,
            const float* __restrict__ nb, const float* __restrict__ hw,
            const float* __restrict__ hb, float* __restrict__ out)
{
  const int b = blockIdx.x;
  const int t = threadIdx.x;
  __shared__ float feat[Dd];
  __shared__ float red[256];
  const float* row = h + (size_t)b * Nn * Dd;   // token 0

  float s = 0.f;
  for (int d = t; d < Dd; d += 256) s += row[d];
  red[t] = s; __syncthreads();
  for (int o = 128; o > 0; o >>= 1) { if (t < o) red[t] += red[t + o]; __syncthreads(); }
  const float mean = red[0] * (1.f / Dd); __syncthreads();

  float v = 0.f;
  for (int d = t; d < Dd; d += 256) { const float x = row[d] - mean; v += x * x; }
  red[t] = v; __syncthreads();
  for (int o = 128; o > 0; o >>= 1) { if (t < o) red[t] += red[t + o]; __syncthreads(); }
  const float rstd = rsqrtf(red[0] * (1.f / Dd) + 1e-6f);

  for (int d = t; d < Dd; d += 256) {
    const float f = (row[d] - mean) * rstd * ns[d] + nb[d];
    feat[d] = f;
    out[Bc * Cc + (size_t)b * Dd + d] = f;
  }
  __syncthreads();
  for (int c = t; c < Cc; c += 256) {
    float a = hb[c];
    for (int d = 0; d < Dd; ++d) a += feat[d] * hw[(size_t)d * Cc + c];
    out[(size_t)b * Cc + c] = a;
  }
}

// ---------------------------------------------------------------------------
static inline int cdiv(int a, int b) { return (a + b - 1) / b; }

extern "C" void kernel_launch(void* const* d_in, const int* in_sizes, int n_in,
                              void* d_out, int out_size, void* d_ws, size_t ws_size,
                              hipStream_t stream)
{
  (void)in_sizes; (void)n_in; (void)out_size; (void)ws_size;

  const float* x        = (const float*)d_in[0];
  const int*   tids     = (const int*)  d_in[1];
  const float* patch_w  = (const float*)d_in[2];
  const float* patch_b  = (const float*)d_in[3];
  const float* cls      = (const float*)d_in[4];
  const float* pos      = (const float*)d_in[5];
  const float* ln1_s    = (const float*)d_in[6];
  const float* ln1_b    = (const float*)d_in[7];
  const float* qkv_w    = (const float*)d_in[8];
  const float* qkv_b    = (const float*)d_in[9];
  const float* proj_w   = (const float*)d_in[10];
  const float* proj_b   = (const float*)d_in[11];
  const float* ln2_s    = (const float*)d_in[12];
  const float* ln2_b    = (const float*)d_in[13];
  const float* fc1_w    = (const float*)d_in[14];
  const float* fc1_b    = (const float*)d_in[15];
  const float* fc2_w    = (const float*)d_in[16];
  const float* fc2_b    = (const float*)d_in[17];
  const float* adw      = (const float*)d_in[18];
  const float* adb      = (const float*)d_in[19];
  const float* auw      = (const float*)d_in[20];
  const float* aub      = (const float*)d_in[21];
  const float* norm_s   = (const float*)d_in[22];
  const float* norm_b   = (const float*)d_in[23];
  const float* head_w   = (const float*)d_in[24];
  const float* head_b   = (const float*)d_in[25];
  float* out = (float*)d_out;

  // ---- workspace layout (~160 MB) ----
  char* w = (char*)d_ws;
  size_t o = 0;
  auto up = [](size_t v) { return (v + 255) & ~(size_t)255; };
  float*  h   = (float*) (w + o); o = up(o + (size_t)BNr * Dd * 4);      // 19.4 MB
  float*  t0  = (float*) (w + o); o = up(o + (size_t)BNr * 4 * Dd * 4);  // 77.5 MB (qkv / fc1)
  float*  t1  = (float*) (w + o); o = up(o + (size_t)BNr * Dd * 4);      // 19.4 MB
  __bf16* abf = (__bf16*)(w + o); o = up(o + (size_t)BNr * 4 * Dd * 2);  // 38.7 MB
  __bf16* wbf = (__bf16*)(w + o); o = up(o + (size_t)Dd * 4 * Dd * 2);   //  4.7 MB

  auto gemm = [&](const __bf16* A, const __bf16* BT, const float* bias,
                  const float* res, float* outp, int M, int N, int K) {
    const int mfull = M / TM;
    if (mfull > 0)
      k_gemm_bf16<false><<<dim3(N / TN, mfull), 256, 0, stream>>>(A, BT, bias, res, outp, M, N, K, 0);
    if (M % TM)
      k_gemm_bf16<true><<<dim3(N / TN, 1), 256, 0, stream>>>(A, BT, bias, res, outp, M, N, K, mfull * TM);
  };
  // fp32 [K][N] weight -> bf16 [N][K] (transposed)
  auto cvtT = [&](const float* src, __bf16* dst, int K, int N) {
    k_f2bfT<<<dim3(N / 32, K / 32), 256, 0, stream>>>(src, dst, K, N);
  };

  // ---- patch embed ----
  cvtT(patch_w, wbf, Dd, Dd);
  k_im2col<<<cdiv(Bc * GG * Dd, 256), 256, 0, stream>>>(x, abf);
  gemm(abf, wbf, patch_b, nullptr, t1, Bc * GG, Dd, Dd);
  k_assemble<<<cdiv(BNr * Dd, 256), 256, 0, stream>>>(t1, cls, pos, h);

  // ---- transformer layers ----
  for (int i = 0; i < 12; ++i) {
    // attention
    k_ln_bf16<<<BNr, 256, 0, stream>>>(h, ln1_s + i * Dd, ln1_b + i * Dd, abf);
    cvtT(qkv_w + (size_t)i * Dd * 3 * Dd, wbf, Dd, 3 * Dd);
    gemm(abf, wbf, qkv_b + i * 3 * Dd, nullptr, t0, BNr, 3 * Dd, Dd);
    k_attn<<<dim3(Nn, Bc), 128, 0, stream>>>(t0, t1);
    k_f2bf<<<cdiv(BNr * Dd, 256), 256, 0, stream>>>(t1, abf, BNr * Dd);
    cvtT(proj_w + (size_t)i * Dd * Dd, wbf, Dd, Dd);
    gemm(abf, wbf, proj_b + i * Dd, h, h, BNr, Dd, Dd);   // h += o@W + b

    // MLP
    k_ln_bf16<<<BNr, 256, 0, stream>>>(h, ln2_s + i * Dd, ln2_b + i * Dd, abf);
    cvtT(fc1_w + (size_t)i * Dd * 4 * Dd, wbf, Dd, 4 * Dd);
    gemm(abf, wbf, fc1_b + i * 4 * Dd, nullptr, t0, BNr, 4 * Dd, Dd);
    k_gelu_bf16<<<cdiv(BNr * 4 * Dd, 256), 256, 0, stream>>>(t0, abf, BNr * 4 * Dd);
    cvtT(fc2_w + (size_t)i * 4 * Dd * Dd, wbf, 4 * Dd, Dd);
    gemm(abf, wbf, fc2_b + i * Dd, nullptr, t1, BNr, Dd, 4 * Dd);

    // residual (+ adapter on first 5 layers)
    if (i < 5) {
      k_adapter<<<BNr, 256, 0, stream>>>(h, t1, tids,
                                         adw + (size_t)i * 10 * Dd * 8,
                                         adb + (size_t)i * 10 * 8,
                                         auw + (size_t)i * 10 * 8 * Dd,
                                         aub + (size_t)i * 10 * Dd);
    } else {
      k_resid<<<cdiv(BNr * Dd, 256), 256, 0, stream>>>(h, t1, BNr * Dd);
    }
  }

  // ---- final norm + head ----
  k_head<<<Bc, 256, 0, stream>>>(h, norm_s, norm_b, head_w, head_b, out);
}
